// MambaGNNLayer_30434138260193
// MI455X (gfx1250) — matrix operand order repaired and checked by reading, not compiled, not run on hardware
//
#include <hip/hip_runtime.h>
#include <hip/hip_bf16.h>

typedef __attribute__((ext_vector_type(16))) _Float16 v16h;
typedef __attribute__((ext_vector_type(8)))  float    v8f;

#define B_   2
#define N_   8192
#define D_   128
#define KNEI 16
#define NS_  16
#define MTOT (B_ * N_)

// ---------------- workspace layout (bytes) ----------------
#define WS_NBR    0                       // int[8192*16]            = 512KB
#define WS_CNT    (WS_NBR + 8192*16*4)    // int[8192]               = 32KB
#define WS_WDT    (WS_CNT + 8192*4)       // f16[128*128] packed [o][i]
#define WS_WB     (WS_WDT + 16384*2)      // f16[16*128]
#define WS_WC     (WS_WB + 2048*2)        // f16[16*128]
#define WS_W1     (WS_WC + 2048*2)        // f16[256*256]
#define WS_W2     (WS_W1 + 65536*2)       // f16[128*256]
#define WS_ANEG   (WS_W2 + 32768*2)       // f32[128*16]
#define WS_AGG    (WS_ANEG + 2048*4)      // f32[16384*128] = 8MB

__device__ __forceinline__ v16h load_bfrag(const _Float16* __restrict__ p) {
  v16h b;
#pragma unroll
  for (int e = 0; e < 8; ++e) { b[e] = p[e]; b[8 + e] = p[16 + e]; }
  return b;
}

#define WMMA(A, Bf, C) \
  __builtin_amdgcn_wmma_f32_16x16x32_f16(false, (A), false, (Bf), (short)0, (C), false, false)

// ---------------- kernel 0: weight prep ----------------
__global__ void prep_kernel(const float* __restrict__ Wdt, const float* __restrict__ WB,
                            const float* __restrict__ WC, const float* __restrict__ W1,
                            const float* __restrict__ W2, const float* __restrict__ Alog,
                            _Float16* __restrict__ WdtT, _Float16* __restrict__ WBT,
                            _Float16* __restrict__ WCT, _Float16* __restrict__ W1T,
                            _Float16* __restrict__ W2T, float* __restrict__ Aneg,
                            float* __restrict__ loss_out) {
  int g = blockIdx.x * blockDim.x + threadIdx.x;
  if (g == 0) *loss_out = 0.0f;
  int idx = g;
  if (idx < 16384) { int o = idx >> 7, i = idx & 127; WdtT[idx] = (_Float16)Wdt[i * 128 + o]; return; }
  idx -= 16384;
  if (idx < 2048)  { int o = idx >> 7, i = idx & 127; WBT[idx]  = (_Float16)WB[i * 16 + o];  return; }
  idx -= 2048;
  if (idx < 2048)  { int o = idx >> 7, i = idx & 127; WCT[idx]  = (_Float16)WC[i * 16 + o];  return; }
  idx -= 2048;
  if (idx < 65536) { int o = idx >> 8, i = idx & 255; W1T[idx]  = (_Float16)W1[i * 256 + o]; return; }
  idx -= 65536;
  if (idx < 32768) { int o = idx >> 8, i = idx & 255; W2T[idx]  = (_Float16)W2[i * 128 + o]; return; }
  idx -= 32768;
  if (idx < 2048)  { Aneg[idx] = -__expf(Alog[idx]); return; }
}

// ---------------- kernel 1: neighbor list extraction ----------------
__global__ __launch_bounds__(256) void nbr_kernel(const float* __restrict__ adj,
                                                  int* __restrict__ nbr,
                                                  int* __restrict__ cnt) {
  __shared__ int s[256];
  int n = blockIdx.x;
  int tid = threadIdx.x;
  const float* row = adj + (size_t)n * N_;
  const int CH = N_ / 256;  // 32 columns per thread, contiguous -> ascending order
  int base_c = tid * CH;
  int lc[KNEI];
  int c_count = 0;
  for (int i = 0; i < CH; ++i) {
    float v = row[base_c + i];
    if (v > 0.0f) {
      if (c_count < KNEI) lc[c_count] = base_c + i;
      c_count++;
    }
  }
  if (tid < KNEI) nbr[(size_t)n * KNEI + tid] = 0;  // default (masked rows)
  s[tid] = c_count;
  __syncthreads();
  // Hillis-Steele inclusive scan over 256 entries
  for (int off = 1; off < 256; off <<= 1) {
    int add = (tid >= off) ? s[tid - off] : 0;
    __syncthreads();
    s[tid] += add;
    __syncthreads();
  }
  int base = s[tid] - c_count;
  int total = s[255];
  if (tid == 0) cnt[n] = (total < KNEI) ? total : KNEI;
  for (int i = 0; i < c_count && i < KNEI; ++i) {
    int p = base + i;
    if (p < KNEI) nbr[(size_t)n * KNEI + p] = lc[i];
  }
}

// ---------------- kernel 2: projections (WMMA) + selective scan ----------------
__global__ __launch_bounds__(64) void scan_kernel(
    const float* __restrict__ feats, const int* __restrict__ nbr_idx,
    const int* __restrict__ cnt_arr, const _Float16* __restrict__ WdtT,
    const _Float16* __restrict__ WBT, const _Float16* __restrict__ WCT,
    const float* __restrict__ b_dt, const float* __restrict__ Aneg,
    const float* __restrict__ Dskip, float* __restrict__ agg) {
  __shared__ float x_lds[2][16][128];
  __shared__ float dt_lds[2][16][128];
  __shared__ float b_lds[2][16][16];
  __shared__ float c_lds[2][16][16];

  int w = threadIdx.x >> 5;
  int lane = threadIdx.x & 31;
  int m = blockIdx.x * 2 + w;
  int bb = m / N_;
  int n = m - bb * N_;
  int r = lane & 15, hi = lane >> 4, col = lane & 15;

  int cnt = cnt_arr[n];
  int nb = nbr_idx[(size_t)n * KNEI + r];
  const float* xrow = feats + ((size_t)bb * N_ + nb) * D_;
  bool valid = (r < cnt);

  // --- gather x-tile: A-fragments + f32 copy to LDS ---
  v16h afrag[4];
#pragma unroll
  for (int ks = 0; ks < 4; ++ks) {
    int c0 = ks * 32 + hi * 8;
    int c1 = ks * 32 + 16 + hi * 8;
#pragma unroll
    for (int e = 0; e < 8; ++e) {
      float v0 = valid ? xrow[c0 + e] : 0.0f;
      float v1 = valid ? xrow[c1 + e] : 0.0f;
      x_lds[w][r][c0 + e] = v0;
      x_lds[w][r][c1 + e] = v1;
      afrag[ks][e]     = (_Float16)v0;
      afrag[ks][8 + e] = (_Float16)v1;
    }
  }

  // --- dt = x @ W_dt : 8 N-tiles x 4 K-steps ---
  v8f dtacc[8] = {};
#pragma unroll
  for (int ks = 0; ks < 4; ++ks) {
#pragma unroll
    for (int j = 0; j < 8; ++j) {
      v16h bf = load_bfrag(WdtT + ((size_t)(j * 16 + col)) * D_ + ks * 32 + hi * 8);
      dtacc[j] = WMMA(afrag[ks], bf, dtacc[j]);
    }
  }
  // --- Bm = x @ W_B, Cm = x @ W_C ---
  v8f bacc = {}, cacc = {};
#pragma unroll
  for (int ks = 0; ks < 4; ++ks) {
    v16h bfB = load_bfrag(WBT + (size_t)col * D_ + ks * 32 + hi * 8);
    bacc = WMMA(afrag[ks], bfB, bacc);
    v16h bfC = load_bfrag(WCT + (size_t)col * D_ + ks * 32 + hi * 8);
    cacc = WMMA(afrag[ks], bfC, cacc);
  }

  // --- softplus(dt + b_dt) -> LDS; Bm, Cm -> LDS ---
#pragma unroll
  for (int j = 0; j < 8; ++j) {
    int d = j * 16 + col;
    float bias = b_dt[d];
#pragma unroll
    for (int v = 0; v < 8; ++v) {
      int t = v + hi * 8;
      float pre = dtacc[j][v] + bias;
      float sp = (pre > 20.0f) ? pre : log1pf(__expf(pre));
      dt_lds[w][t][d] = sp;
    }
  }
#pragma unroll
  for (int v = 0; v < 8; ++v) {
    int t = v + hi * 8;
    b_lds[w][t][col] = bacc[v];
    c_lds[w][t][col] = cacc[v];
  }
  __syncthreads();

  // --- S6 recurrence: each lane owns d = lane + 32*i ---
  float A_[4][16];
  float h[4][16];
  float Dv[4];
  float y[4] = {0.f, 0.f, 0.f, 0.f};
  int dch[4];
#pragma unroll
  for (int i = 0; i < 4; ++i) {
    dch[i] = lane + 32 * i;
    Dv[i] = Dskip[dch[i]];
#pragma unroll
    for (int ns = 0; ns < NS_; ++ns) {
      A_[i][ns] = Aneg[dch[i] * NS_ + ns];
      h[i][ns] = 0.0f;
    }
  }
  int last = (cnt > 0) ? (cnt - 1) : 0;
  for (int t = 0; t < KNEI; ++t) {
    if (t >= cnt) break;  // cnt uniform across the wave
    float Bt[16], Ct[16];
#pragma unroll
    for (int ns = 0; ns < NS_; ++ns) {
      Bt[ns] = b_lds[w][t][ns];
      Ct[ns] = c_lds[w][t][ns];
    }
#pragma unroll
    for (int i = 0; i < 4; ++i) {
      float dtv = dt_lds[w][t][dch[i]];
      float xv = x_lds[w][t][dch[i]];
      float dbx = dtv * xv;
#pragma unroll
      for (int ns = 0; ns < NS_; ++ns)
        h[i][ns] = __expf(dtv * A_[i][ns]) * h[i][ns] + dbx * Bt[ns];
      if (t == last) {
        float acc = Dv[i] * xv;
#pragma unroll
        for (int ns = 0; ns < NS_; ++ns) acc += h[i][ns] * Ct[ns];
        y[i] = acc;
      }
    }
  }
#pragma unroll
  for (int i = 0; i < 4; ++i) agg[(size_t)m * D_ + dch[i]] = y[i];
}

// ---------------- kernel 3: MLP (2x WMMA GEMM) + residual + LayerNorm ----------------
__global__ __launch_bounds__(64) void mlp_kernel(
    const float* __restrict__ feats, const float* __restrict__ agg,
    const _Float16* __restrict__ W1T, const float* __restrict__ b1,
    const _Float16* __restrict__ W2T, const float* __restrict__ b2,
    const float* __restrict__ gamma, const float* __restrict__ beta,
    float* __restrict__ out) {
  __shared__ _Float16 h_lds[2][16][256];
  __shared__ float z_lds[2][16][128];

  int w = threadIdx.x >> 5;
  int lane = threadIdx.x & 31;
  int tile = blockIdx.x * 2 + w;
  int r = lane & 15, hi = lane >> 4, col = lane & 15;
  size_t mrow = (size_t)tile * 16 + r;
  const float* frow = feats + mrow * D_;
  const float* arow = agg + mrow * D_;

  // --- H = relu([feat, agg] @ W1 + b1) : 16 N-tiles x 8 K-steps ---
  v8f hacc[16] = {};
#pragma unroll
  for (int ks = 0; ks < 8; ++ks) {
    v16h af;
    int c0 = ks * 32 + hi * 8;
    int c1 = ks * 32 + 16 + hi * 8;
#pragma unroll
    for (int e = 0; e < 8; ++e) {
      af[e]     = (c0 + e < D_) ? (_Float16)frow[c0 + e] : (_Float16)arow[c0 + e - D_];
      af[8 + e] = (c1 + e < D_) ? (_Float16)frow[c1 + e] : (_Float16)arow[c1 + e - D_];
    }
#pragma unroll
    for (int j = 0; j < 16; ++j) {
      v16h bf = load_bfrag(W1T + ((size_t)(j * 16 + col)) * 256 + ks * 32 + hi * 8);
      hacc[j] = WMMA(af, bf, hacc[j]);
    }
  }
#pragma unroll
  for (int j = 0; j < 16; ++j) {
    int c = j * 16 + col;
    float bias = b1[c];
#pragma unroll
    for (int v = 0; v < 8; ++v) {
      int t = v + hi * 8;
      float hv = hacc[j][v] + bias;
      h_lds[w][t][c] = (_Float16)(hv > 0.0f ? hv : 0.0f);
    }
  }
  __syncthreads();

  // --- update = H @ W2 + b2 : 8 N-tiles x 8 K-steps (A from LDS) ---
  v8f uacc[8] = {};
#pragma unroll
  for (int ks = 0; ks < 8; ++ks) {
    v16h af;
    int c0 = ks * 32 + hi * 8;
    int c1 = ks * 32 + 16 + hi * 8;
#pragma unroll
    for (int e = 0; e < 8; ++e) {
      af[e]     = h_lds[w][r][c0 + e];
      af[8 + e] = h_lds[w][r][c1 + e];
    }
#pragma unroll
    for (int j = 0; j < 8; ++j) {
      v16h bf = load_bfrag(W2T + ((size_t)(j * 16 + col)) * 256 + ks * 32 + hi * 8);
      uacc[j] = WMMA(af, bf, uacc[j]);
    }
  }
  // --- z = feat + update + b2 ---
#pragma unroll
  for (int j = 0; j < 8; ++j) {
    int d = j * 16 + col;
    float bias = b2[d];
#pragma unroll
    for (int v = 0; v < 8; ++v) {
      int t = v + hi * 8;
      float z = feats[((size_t)tile * 16 + t) * D_ + d] + uacc[j][v] + bias;
      z_lds[w][t][d] = z;
    }
  }
  __syncthreads();

  // --- LayerNorm: lane pair (l, l^16) handles row r ---
  float s = 0.0f, ss = 0.0f;
#pragma unroll
  for (int k = 0; k < 64; ++k) {
    float v = z_lds[w][r][hi * 64 + k];
    s += v;
    ss += v * v;
  }
  s  += __shfl_xor(s, 16, 32);
  ss += __shfl_xor(ss, 16, 32);
  float mu = s * (1.0f / 128.0f);
  float var = ss * (1.0f / 128.0f) - mu * mu;
  float rstd = rsqrtf(var + 1e-5f);
  float* orow = out + ((size_t)tile * 16 + r) * D_;
#pragma unroll
  for (int k = 0; k < 64; ++k) {
    int d = hi * 64 + k;
    float v = z_lds[w][r][d];
    orow[d] = (v - mu) * rstd * gamma[d] + beta[d];
  }
}

// ---------------- launch ----------------
extern "C" void kernel_launch(void* const* d_in, const int* in_sizes, int n_in,
                              void* d_out, int out_size, void* d_ws, size_t ws_size,
                              hipStream_t stream) {
  (void)in_sizes; (void)n_in; (void)ws_size;
  const float* feats = (const float*)d_in[0];
  const float* adj   = (const float*)d_in[1];
  const float* W_dt  = (const float*)d_in[2];
  const float* b_dt  = (const float*)d_in[3];
  const float* W_B   = (const float*)d_in[4];
  const float* W_C   = (const float*)d_in[5];
  const float* A_log = (const float*)d_in[6];
  const float* Dskip = (const float*)d_in[7];
  const float* W1    = (const float*)d_in[8];
  const float* b1    = (const float*)d_in[9];
  const float* W2    = (const float*)d_in[10];
  const float* b2    = (const float*)d_in[11];
  const float* gamma = (const float*)d_in[12];
  const float* beta  = (const float*)d_in[13];
  float* out = (float*)d_out;

  char* ws = (char*)d_ws;
  int*       nbr   = (int*)(ws + WS_NBR);
  int*       cnt   = (int*)(ws + WS_CNT);
  _Float16*  WdtT  = (_Float16*)(ws + WS_WDT);
  _Float16*  WBT   = (_Float16*)(ws + WS_WB);
  _Float16*  WCT   = (_Float16*)(ws + WS_WC);
  _Float16*  W1T   = (_Float16*)(ws + WS_W1);
  _Float16*  W2T   = (_Float16*)(ws + WS_W2);
  float*     Aneg  = (float*)(ws + WS_ANEG);
  float*     agg   = (float*)(ws + WS_AGG);

  // 0) weight prep (+ cons_loss = 0 at tail of d_out)
  int prep_elems = 16384 + 2048 + 2048 + 65536 + 32768 + 2048;
  prep_kernel<<<(prep_elems + 255) / 256, 256, 0, stream>>>(
      W_dt, W_B, W_C, W1, W2, A_log, WdtT, WBT, WCT, W1T, W2T, Aneg,
      out + (out_size - 1));

  // 1) neighbor extraction
  nbr_kernel<<<N_, 256, 0, stream>>>(adj, nbr, cnt);

  // 2) WMMA projections + selective scan (1 wave / node, 2 waves / block)
  scan_kernel<<<MTOT / 2, 64, 0, stream>>>(feats, nbr, cnt, WdtT, WBT, WCT,
                                           b_dt, Aneg, Dskip, agg);

  // 3) WMMA MLP + residual + LayerNorm (1 wave / 16 nodes)
  mlp_kernel<<<(MTOT / 16) / 2, 64, 0, stream>>>(feats, agg, W1T, b1, W2T, b2,
                                                 gamma, beta, out);
}
